// SparseNeighborhoodAttentionBlock_34711925686576
// MI455X (gfx1250) — matrix operand order repaired and checked by reading, not compiled, not run on hardware
//
#include <hip/hip_runtime.h>

// ---------------------------------------------------------------------------
// Sparse neighborhood attention block for MI455X (gfx1250, wave32, WMMA).
// Heavy GEMMs on v_wmma_f32_16x16x32_bf16; gather via async-to-LDS DMA path
// (global_load_async_to_lds_b128 + s_wait_asynccnt); softmax/RoPE in VALU f32.
// ---------------------------------------------------------------------------

typedef __attribute__((ext_vector_type(16))) __bf16 v16bf;
typedef __attribute__((ext_vector_type(8)))  float  v8f;

#define EMBED   256
#define HEADS   8
#define HD      32
#define NQ      2048
#define KPQ     164
#define KCHUNK  16
#define NCHUNK  11          // ceil(164/16)
#define QPB     2           // queries per workgroup (fused kernel)
#define ROWS    (QPB*KCHUNK) // 32 gathered rows per chunk
#define INV_SQRT_HD 0.17677669529663689f

// ---- workspace layout (bytes) ----
#define WS_XB   524288      // x_bf16   (2048*256 u16)
#define WS_QROT 1572864     // q_rot    (2048*256 f32)
#define WS_ATTN 3670016     // attn     (2048*256 u16)

__device__ __forceinline__ unsigned short f2bf(float f) {
  union { float f; unsigned u; } c; c.f = f;
  unsigned r = c.u + 0x7FFFu + ((c.u >> 16) & 1u);   // round-to-nearest-even
  return (unsigned short)(r >> 16);
}
__device__ __forceinline__ float bf2f(unsigned short u) {
  union { unsigned u; float f; } c; c.u = ((unsigned)u) << 16;
  return c.f;
}

__device__ __forceinline__ v8f wmma_bf16(v16bf a, v16bf b, v8f c) {
  return __builtin_amdgcn_wmma_f32_16x16x32_bf16(false, a, false, b,
                                                 (short)0, c, false, false);
}

// A fragment (16x32 bf16, row-major src, stride lda)
__device__ __forceinline__ v16bf load_frag_a(const unsigned short* A, int lda,
                                             int M0, int K0, int lane) {
  int half = lane >> 4, r = lane & 15;
  const unsigned short* p = A + (M0 + r) * lda + K0 + half * 8;
  union { uint4 u[2]; v16bf v; } u;
  u.u[0] = *(const uint4*)p;
  u.u[1] = *(const uint4*)(p + 16);
  return u.v;
}

// B fragment (32x16 bf16) from TRANSPOSED weights (N x K row-major)
__device__ __forceinline__ v16bf load_frag_b(const unsigned short* BT, int ldb,
                                             int N0, int K0, int lane) {
  int half = lane >> 4, n = lane & 15;
  const unsigned short* p = BT + (N0 + n) * ldb + K0 + half * 8;
  union { uint4 u[2]; v16bf v; } u;
  u.u[0] = *(const uint4*)p;
  u.u[1] = *(const uint4*)(p + 16);
  return u.v;
}

// D (16x16 f32) -> bf16 LDS tile, stride EMBED
__device__ __forceinline__ void store_d_lds_bf16(unsigned short* D, int M0,
                                                 int N0, v8f acc, int lane) {
  int col = N0 + (lane & 15);
  int mb  = M0 + ((lane >> 4) << 3);
#pragma unroll
  for (int r = 0; r < 8; r++) D[(mb + r) * EMBED + col] = f2bf(acc[r]);
}

// ---------------------------------------------------------------------------
// Kernel 1: convert 4 weight matrices f32 -> bf16, transposed (out: N x K).
// ---------------------------------------------------------------------------
__global__ void convert_weights_kernel(const float* __restrict__ qw,
                                       const float* __restrict__ kw,
                                       const float* __restrict__ vw,
                                       const float* __restrict__ ow,
                                       unsigned short* __restrict__ wT) {
  int t = blockIdx.x * blockDim.x + threadIdx.x;   // 0..262143
  int m = t >> 16;
  int e = t & 65535;
  int n = e >> 8, k = e & 255;
  const float* src = (m == 0) ? qw : (m == 1) ? kw : (m == 2) ? vw : ow;
  wT[m * 65536 + n * 256 + k] = f2bf(src[k * 256 + n]);
}

// ---------------------------------------------------------------------------
// Kernel 2: LayerNorm -> bf16 activations. One wave32 per row.
// ---------------------------------------------------------------------------
__global__ __launch_bounds__(256)
void layernorm_kernel(const float* __restrict__ q,
                      const float* __restrict__ w,
                      const float* __restrict__ b,
                      unsigned short* __restrict__ xb) {
  int lane = threadIdx.x & 31, wave = threadIdx.x >> 5;
  int row = blockIdx.x * 8 + wave;
  const float* src = q + row * EMBED;
  float vals[8]; float s = 0.f, s2 = 0.f;
#pragma unroll
  for (int i = 0; i < 8; i++) {
    float v = src[lane + i * 32];
    vals[i] = v; s += v; s2 += v * v;
  }
  for (int m = 16; m >= 1; m >>= 1) {
    s  += __shfl_xor(s,  m, 32);
    s2 += __shfl_xor(s2, m, 32);
  }
  float mu  = s * (1.f / EMBED);
  float var = s2 * (1.f / EMBED) - mu * mu;
  float inv = rsqrtf(var + 1e-5f);
#pragma unroll
  for (int i = 0; i < 8; i++) {
    int c = lane + i * 32;
    xb[row * EMBED + c] = f2bf((vals[i] - mu) * inv * w[c] + b[c]);
  }
}

// ---------------------------------------------------------------------------
// Kernel 3: q = x @ q_w via WMMA, fused RoPE on D fragments -> q_rot f32.
// ---------------------------------------------------------------------------
__global__ __launch_bounds__(128)
void qproj_rope_kernel(const unsigned short* __restrict__ xb,
                       const unsigned short* __restrict__ qwT,
                       const float* __restrict__ qsp,
                       const float* __restrict__ rope,
                       float* __restrict__ qrot) {
  int lane = threadIdx.x & 31, wave = threadIdx.x >> 5;
  int tile = blockIdx.x * 4 + wave;         // 2048 tiles: 128 M x 16 N
  int M0 = (tile >> 4) << 4;
  int N0 = (tile & 15) << 4;
  v8f acc = {};
  for (int K0 = 0; K0 < EMBED; K0 += 32) {
    v16bf a  = load_frag_a(xb, EMBED, M0, K0, lane);
    v16bf bm = load_frag_b(qwT, EMBED, N0, K0, lane);
    acc = wmma_bf16(a, bm, acc);
  }
  int c  = N0 + (lane & 15);
  int ph = (c >> 1) & 15;
  float f0 = rope[ph], f1 = rope[16 + ph];
  bool even = (c & 1) == 0;
  int mbase = M0 + ((lane >> 4) << 3);
#pragma unroll
  for (int r = 0; r < 8; r++) {
    int m = mbase + r;
    float ang = qsp[2 * m] * f0 + qsp[2 * m + 1] * f1;   // level coord = 0
    float cs = __cosf(ang), sn = __sinf(ang);
    float a0 = acc[r];
    float bp = __shfl_xor(a0, 1, 32);
    float x1 = even ? a0 : bp;
    float x2 = even ? bp : a0;
    qrot[m * EMBED + c] = even ? (x1 * cs - x2 * sn) : (x1 * sn + x2 * cs);
  }
}

// ---------------------------------------------------------------------------
// Kernel 4: fused gather (async-to-LDS) + k/v WMMA + RoPE(k) + online softmax.
// 256 threads (8 waves) handle QPB=2 queries; wave h <-> head h (both queries).
// LDS pool aliasing (disjoint lifetimes):
//   phase 2: pool[0..32K)  = 32 rows of f32 gather staging
//   phase 3+: pool[0..16K) = feats bf16 (32x256), pool[16K..32K) = k bf16
// ---------------------------------------------------------------------------
__global__ __launch_bounds__(256)
void fused_attn_kernel(const float* __restrict__ qsp,
                       const float* __restrict__ sfm,
                       const unsigned short* __restrict__ kwT,
                       const unsigned short* __restrict__ vwT,
                       const float* __restrict__ rope,
                       const int* __restrict__ qbo,
                       const float* __restrict__ qrot,
                       unsigned short* __restrict__ attnb) {
  __shared__ __align__(16) unsigned char pool_s[ROWS * EMBED * 4]; // 32 KB
  __shared__ __align__(16) unsigned short v_s[ROWS * EMBED];       // 16 KB
  __shared__ float q_s[QPB * EMBED];                               // 2 KB
  __shared__ float kpx_s[ROWS], kpy_s[ROWS], klv_s[ROWS];
  __shared__ int   valid_s[ROWS];
  __shared__ unsigned long long koff_s[ROWS];

  float*          stage_f = (float*)pool_s;
  unsigned short* feats_s = (unsigned short*)pool_s;
  unsigned short* k_s     = (unsigned short*)(pool_s + ROWS * EMBED * 2);

  int t = threadIdx.x, lane = t & 31, wave = t >> 5;
  int qi0 = blockIdx.x * QPB;

  float pxq[QPB], pyq[QPB];
  int bq[QPB];
#pragma unroll
  for (int qq = 0; qq < QPB; qq++) {
    int qi = qi0 + qq;
    pxq[qq] = qsp[2 * qi]; pyq[qq] = qsp[2 * qi + 1];
    int b = -1;
#pragma unroll
    for (int j = 0; j < 3; j++) b += (qbo[j] <= qi) ? 1 : 0;
    bq[qq] = b;
  }
#pragma unroll
  for (int qq = 0; qq < QPB; qq++)
    q_s[qq * EMBED + t] = qrot[(qi0 + qq) * EMBED + t];

  int h = wave;                      // wave <-> head
  float m_run[QPB], l_run[QPB], accv[QPB];
#pragma unroll
  for (int qq = 0; qq < QPB; qq++) { m_run[qq] = -3.0e38f; l_run[qq] = 0.f; accv[qq] = 0.f; }

  for (int cc = 0; cc < NCHUNK; cc++) {
    __syncthreads();
    // --- phase 1: per-key metadata (threads 0..31: qq = t>>4, key = t&15) ---
    if (t < ROWS) {
      int qq = t >> 4;
      int kk = cc * KCHUNK + (t & 15);
      int l, base, s;
      if (kk < 9)       { l = 0; base = 0;  s = 3; }
      else if (kk < 34) { l = 1; base = 9;  s = 5; }
      else if (kk < 83) { l = 2; base = 34; s = 7; }
      else              { l = 3; base = 83; s = 9; }
      int loc = kk - base;
      int di = loc / s - (s - 1) / 2;
      int dj = loc % s - (s - 1) / 2;
      float scale = 1.0f / (float)(1 << l);
      int i0 = (int)floorf(pxq[qq] * scale) + di;
      int i1 = (int)floorf(pyq[qq] * scale) + dj;
      int sh = 128 >> l;
      int ok = (kk < KPQ) && (i0 >= 0) && (i0 < sh) && (i1 >= 0) && (i1 < sh);
      int i0c = min(max(i0, 0), sh - 1);
      int i1c = min(max(i1, 0), sh - 1);
      valid_s[t] = ok;
      koff_s[t]  = ((((unsigned long long)bq[qq] * 4 + l) * 128 + i0c) * 128 + i1c)
                   * (unsigned long long)EMBED;
      float str = (float)(1 << l);
      kpx_s[t] = ((float)i0 + 0.5f) * str;
      kpy_s[t] = ((float)i1 + 0.5f) * str;
      klv_s[t] = (float)l;
    }
    __syncthreads();
    // --- phase 2a: async DMA gather 32 rows (f32) into LDS staging ---
    {
      int r = t >> 3, sub = t & 7;                 // 8 threads x 128B per row
      const float* gp = sfm + koff_s[r] + sub * 32;
      unsigned lds0 = (unsigned)(uintptr_t)(stage_f + r * EMBED + sub * 32);
      unsigned long long ga = (unsigned long long)(uintptr_t)gp;
#pragma unroll
      for (int i = 0; i < 8; i++) {
        asm volatile("global_load_async_to_lds_b128 %0, %1, off offset:%2"
                     :: "v"(lds0), "v"(ga), "i"(i * 16) : "memory");
      }
      asm volatile("s_wait_asynccnt 0" ::: "memory");
    }
    __syncthreads();
    // --- phase 2b: f32 -> bf16 in place (read regs, barrier, write) ---
    {
      int seg = t & 15;
      float tmp[2][16];
#pragma unroll
      for (int it = 0; it < 2; it++) {
        int r = (t >> 4) + 16 * it;
        int ok = valid_s[r];
        const float* src = stage_f + r * EMBED + seg * 16;
#pragma unroll
        for (int i = 0; i < 16; i++) tmp[it][i] = ok ? src[i] : 0.f;
      }
      __syncthreads();
#pragma unroll
      for (int it = 0; it < 2; it++) {
        int r = (t >> 4) + 16 * it;
        union { unsigned short us[16]; uint4 u4[2]; } o;
#pragma unroll
        for (int i = 0; i < 16; i++) o.us[i] = f2bf(tmp[it][i]);
        *(uint4*)&feats_s[r * EMBED + seg * 16]     = o.u4[0];
        *(uint4*)&feats_s[r * EMBED + seg * 16 + 8] = o.u4[1];
      }
    }
    __syncthreads();
    // --- phase 3: k/v = feats @ w (WMMA); B frags reused over 2 M-tiles ---
    {
      int N0 = wave * 32;
      v8f ak[QPB][2], av[QPB][2];
#pragma unroll
      for (int m = 0; m < QPB; m++)
        for (int n = 0; n < 2; n++) { ak[m][n] = (v8f){}; av[m][n] = (v8f){}; }
      for (int K0 = 0; K0 < EMBED; K0 += 32) {
        v16bf a0  = load_frag_a(feats_s, EMBED, 0,  K0, lane);
        v16bf a1  = load_frag_a(feats_s, EMBED, 16, K0, lane);
        v16bf bk0 = load_frag_b(kwT, EMBED, N0,      K0, lane);
        v16bf bk1 = load_frag_b(kwT, EMBED, N0 + 16, K0, lane);
        v16bf bv0 = load_frag_b(vwT, EMBED, N0,      K0, lane);
        v16bf bv1 = load_frag_b(vwT, EMBED, N0 + 16, K0, lane);
        ak[0][0] = wmma_bf16(a0, bk0, ak[0][0]);
        ak[0][1] = wmma_bf16(a0, bk1, ak[0][1]);
        ak[1][0] = wmma_bf16(a1, bk0, ak[1][0]);
        ak[1][1] = wmma_bf16(a1, bk1, ak[1][1]);
        av[0][0] = wmma_bf16(a0, bv0, av[0][0]);
        av[0][1] = wmma_bf16(a0, bv1, av[0][1]);
        av[1][0] = wmma_bf16(a1, bv0, av[1][0]);
        av[1][1] = wmma_bf16(a1, bv1, av[1][1]);
      }
#pragma unroll
      for (int m = 0; m < QPB; m++) {
        store_d_lds_bf16(k_s, m * 16, N0,      ak[m][0], lane);
        store_d_lds_bf16(k_s, m * 16, N0 + 16, ak[m][1], lane);
        store_d_lds_bf16(v_s, m * 16, N0,      av[m][0], lane);
        store_d_lds_bf16(v_s, m * 16, N0 + 16, av[m][1], lane);
      }
    }
    __syncthreads();
    // --- phase 4: RoPE on k (32 keys x 128 pairs, 16 pairs/thread) ---
#pragma unroll
    for (int i = 0; i < 16; i++) {
      int pp  = t + i * 256;
      int key = pp >> 7;
      int pin = pp & 127;
      int p   = pin & 15;
      int col = ((pin >> 4) << 5) + (p << 1);
      float ang = kpx_s[key] * rope[p] + kpy_s[key] * rope[16 + p]
                + klv_s[key] * rope[32 + p];
      float cs = __cosf(ang), sn = __sinf(ang);
      float x1 = bf2f(k_s[key * EMBED + col]);
      float x2 = bf2f(k_s[key * EMBED + col + 1]);
      k_s[key * EMBED + col]     = f2bf(x1 * cs - x2 * sn);
      k_s[key * EMBED + col + 1] = f2bf(x1 * sn + x2 * cs);
    }
    __syncthreads();
    // --- phase 5: logits + online softmax (wave h == head h, both queries) ---
#pragma unroll
    for (int qq = 0; qq < QPB; qq++) {
      int key = lane & 15;                  // lanes 16-31 mirror 0-15
      int row = qq * KCHUNK + key;
      float dot = 0.f;
#pragma unroll
      for (int d = 0; d < HD; d++)
        dot += q_s[qq * EMBED + h * HD + d] * bf2f(k_s[row * EMBED + h * HD + d]);
      float logit = valid_s[row] ? dot * INV_SQRT_HD : -1.0e9f;
      float cmax = logit;
      for (int mm = 8; mm >= 1; mm >>= 1)
        cmax = fmaxf(cmax, __shfl_xor(cmax, mm, 32));
      float mnew = fmaxf(m_run[qq], cmax);
      float sc   = __expf(m_run[qq] - mnew);
      float pr   = __expf(logit - mnew);
      float psum = pr;
      for (int mm = 8; mm >= 1; mm >>= 1)
        psum += __shfl_xor(psum, mm, 32);
      l_run[qq] = l_run[qq] * sc + psum;
      m_run[qq] = mnew;
      accv[qq] *= sc;                        // lane == output dim d
#pragma unroll
      for (int j = 0; j < KCHUNK; j++) {
        float pj = __shfl(pr, j, 32);
        accv[qq] += pj * bf2f(v_s[(qq * KCHUNK + j) * EMBED + h * HD + lane]);
      }
    }
  }
#pragma unroll
  for (int qq = 0; qq < QPB; qq++)
    attnb[(qi0 + qq) * EMBED + h * HD + lane] = f2bf(accv[qq] / l_run[qq]);
}

// ---------------------------------------------------------------------------
// Kernel 5: out = attn @ out_w + residual (WMMA), f32 output.
// ---------------------------------------------------------------------------
__global__ __launch_bounds__(128)
void outproj_kernel(const unsigned short* __restrict__ attnb,
                    const unsigned short* __restrict__ owT,
                    const float* __restrict__ residual,
                    float* __restrict__ out) {
  int lane = threadIdx.x & 31, wave = threadIdx.x >> 5;
  int tile = blockIdx.x * 4 + wave;
  int M0 = (tile >> 4) << 4;
  int N0 = (tile & 15) << 4;
  v8f acc = {};
  for (int K0 = 0; K0 < EMBED; K0 += 32) {
    v16bf a  = load_frag_a(attnb, EMBED, M0, K0, lane);
    v16bf bm = load_frag_b(owT, EMBED, N0, K0, lane);
    acc = wmma_bf16(a, bm, acc);
  }
  int col = N0 + (lane & 15);
  int mb  = M0 + ((lane >> 4) << 3);
#pragma unroll
  for (int r = 0; r < 8; r++)
    out[(mb + r) * EMBED + col] = acc[r] + residual[(mb + r) * EMBED + col];
}

// ---------------------------------------------------------------------------
extern "C" void kernel_launch(void* const* d_in, const int* in_sizes, int n_in,
                              void* d_out, int out_size, void* d_ws, size_t ws_size,
                              hipStream_t stream) {
  const float* query = (const float*)d_in[0];
  const float* qsp   = (const float*)d_in[1];
  const float* sfm   = (const float*)d_in[2];
  const float* nw    = (const float*)d_in[3];
  const float* nb    = (const float*)d_in[4];
  const float* q_w   = (const float*)d_in[5];
  const float* k_w   = (const float*)d_in[6];
  const float* v_w   = (const float*)d_in[7];
  const float* o_w   = (const float*)d_in[8];
  const float* rope  = (const float*)d_in[9];
  const int*   qbo   = (const int*)d_in[10];
  // d_in[11] = level_spatial_shapes (derived analytically: 128 >> level)

  char* ws = (char*)d_ws;
  unsigned short* wT    = (unsigned short*)ws;              // 4 x 65536 u16
  unsigned short* xb    = (unsigned short*)(ws + WS_XB);
  float*          qrot  = (float*)(ws + WS_QROT);
  unsigned short* attnb = (unsigned short*)(ws + WS_ATTN);
  float* out = (float*)d_out;

  convert_weights_kernel<<<1024, 256, 0, stream>>>(q_w, k_w, v_w, o_w, wT);
  layernorm_kernel<<<NQ / 8, 256, 0, stream>>>(query, nw, nb, xb);
  qproj_rope_kernel<<<512, 128, 0, stream>>>(xb, wT, qsp, rope, qrot);
  fused_attn_kernel<<<NQ / QPB, 256, 0, stream>>>(qsp, sfm, wT + 65536,
                                                  wT + 131072, rope, qbo,
                                                  qrot, attnb);
  outproj_kernel<<<512, 128, 0, stream>>>(attnb, wT + 3 * 65536, query, out);
}
